// GRUModulatedConv2d_63136019251953
// MI455X (gfx1250) — matrix operand order repaired and checked by reading, not compiled, not run on hardware
//
#include <hip/hip_runtime.h>
#include <hip/hip_bf16.h>
#include <math.h>

// ---------- types ----------
typedef __attribute__((ext_vector_type(16))) __bf16        v16bf;
typedef __attribute__((ext_vector_type(8)))  float          v8f;
typedef __attribute__((ext_vector_type(8)))  unsigned short v8us;
typedef __attribute__((ext_vector_type(4)))  int            v4i;

typedef __attribute__((address_space(1))) v4i* gv4i_p;   // global
typedef __attribute__((address_space(3))) v4i* lv4i_p;   // LDS

#define BATCH 8
#define CINC  512
#define COUTC 512
#define HIDC  512
#define HWDIM 64
#define XPAD  66   // 64 + 2 halo

#if __has_builtin(__builtin_amdgcn_global_load_async_to_lds_b128) && \
    __has_builtin(__builtin_amdgcn_s_wait_asynccnt)
#define HAVE_ASYNC 1
#else
#define HAVE_ASYNC 0
#endif

// round-to-nearest-even float -> bf16 bits
__device__ __forceinline__ unsigned short f2bf(float f) {
  unsigned int u = __float_as_uint(f);
  unsigned int r = (u + 0x7FFFu + ((u >> 16) & 1u)) >> 16;
  return (unsigned short)r;
}

// 16B global -> LDS copy (async on CDNA5, sync fallback otherwise)
__device__ __forceinline__ void copy16(unsigned short* ldst, const unsigned short* gsrc) {
#if HAVE_ASYNC
  __builtin_amdgcn_global_load_async_to_lds_b128(
      (gv4i_p)(gsrc), (lv4i_p)(ldst), 0, 0);
#else
  *(v8us*)ldst = *(const v8us*)gsrc;
#endif
}

// ---------------- GRU gates: gx = x@Wih^T + bih ; gh = h@Whh^T + bhh ----------------
__global__ void k_gates(const float* __restrict__ x, const float* __restrict__ h,
                        const float* __restrict__ w_ih, const float* __restrict__ w_hh,
                        const float* __restrict__ b_ih, const float* __restrict__ b_hh,
                        float* __restrict__ gx, float* __restrict__ gh) {
  int idx = blockIdx.x * blockDim.x + threadIdx.x;   // 8 * 1536
  if (idx >= BATCH * 3 * HIDC) return;
  int b = idx & 7, j = idx >> 3;
  const float* wi = w_ih + j * HIDC;
  const float* wh = w_hh + j * HIDC;
  const float* xb = x + b * HIDC;
  const float* hb = h + b * HIDC;
  float a0 = 0.f, a1 = 0.f;
  for (int k = 0; k < HIDC; k++) { a0 += xb[k] * wi[k]; a1 += hb[k] * wh[k]; }
  gx[b * 3 * HIDC + j] = a0 + b_ih[j];
  gh[b * 3 * HIDC + j] = a1 + b_hh[j];
}

// ---------------- GRU combine -> h2 ----------------
__global__ void k_h2(const float* __restrict__ gx, const float* __restrict__ gh,
                     const float* __restrict__ h1, float* __restrict__ h2out) {
  int idx = blockIdx.x * blockDim.x + threadIdx.x;   // 4096
  if (idx >= BATCH * HIDC) return;
  int b = idx >> 9, i = idx & 511;
  int base = b * 3 * HIDC;
  float xr = gx[base + i],            hr = gh[base + i];
  float xz = gx[base + HIDC + i],     hz = gh[base + HIDC + i];
  float xn = gx[base + 2 * HIDC + i], hn = gh[base + 2 * HIDC + i];
  float r = 1.f / (1.f + expf(-(xr + hr)));
  float z = 1.f / (1.f + expf(-(xz + hz)));
  float n = tanhf(xn + r * hn);
  h2out[idx] = (1.f - z) * n + z * h1[idx];
}

// ---------------- s = h2 @ (aff_w*scale)^T + aff_b ----------------
__global__ void k_s(const float* __restrict__ h2, const float* __restrict__ aff_w,
                    const float* __restrict__ aff_b, float* __restrict__ s) {
  int idx = blockIdx.x * blockDim.x + threadIdx.x;   // 4096
  if (idx >= BATCH * CINC) return;
  int b = idx & 7, c = idx >> 3;
  const float* hv = h2 + b * HIDC;
  const float* wv = aff_w + c * HIDC;
  float a = 0.f;
  for (int k = 0; k < HIDC; k++) a += hv[k] * wv[k];
  s[b * CINC + c] = a * 0.04419417382415922f + aff_b[c];   // 1/sqrt(512)
}

// ---------------- style = sqrt(softmax(s, axis=1)) ----------------
__global__ void k_style(const float* __restrict__ s, float* __restrict__ style) {
  __shared__ float red[256];
  int b = blockIdx.x, t = threadIdx.x;
  float v0 = s[b * CINC + t];
  float v1 = s[b * CINC + 256 + t];
  red[t] = fmaxf(v0, v1);
  __syncthreads();
  for (int o = 128; o > 0; o >>= 1) { if (t < o) red[t] = fmaxf(red[t], red[t + o]); __syncthreads(); }
  float mx = red[0];
  __syncthreads();
  float e0 = expf(v0 - mx), e1 = expf(v1 - mx);
  red[t] = e0 + e1;
  __syncthreads();
  for (int o = 128; o > 0; o >>= 1) { if (t < o) red[t] += red[t + o]; __syncthreads(); }
  float inv = 1.f / red[0];
  style[b * CINC + t]       = sqrtf(e0 * inv);
  style[b * CINC + 256 + t] = sqrtf(e1 * inv);
}

// ---------------- demodulated weight -> bf16, layout wq[tap][cout][cin] ----------------
__global__ void k_wprep(const float* __restrict__ cw, unsigned short* __restrict__ wq) {
  int idx = blockIdx.x * blockDim.x + threadIdx.x;   // 512*512
  if (idx >= COUTC * CINC) return;
  int co = idx >> 9, ci = idx & 511;
  const float* w = cw + (co * CINC + ci) * 9;
  float wv[9];
  float ss = 0.f;
#pragma unroll
  for (int k = 0; k < 9; k++) { wv[k] = w[k]; ss += wv[k] * wv[k]; }
  float dn = rsqrtf(ss + 1e-8f);
#pragma unroll
  for (int k = 0; k < 9; k++)
    wq[k * (COUTC * CINC) + co * CINC + ci] = f2bf(wv[k] * dn);
}

// ---------------- zero the padded styled-input buffer (halo) ----------------
__global__ void k_xzero(unsigned int* __restrict__ xs32, int n32) {
  int idx = blockIdx.x * blockDim.x + threadIdx.x;
  if (idx < n32) xs32[idx] = 0u;
}

// ------- styled input -> bf16, transposed to xs[b][y+1][x+1][ci] (zero halo) -------
__global__ void k_xprep(const float* __restrict__ input, const float* __restrict__ style,
                        unsigned short* __restrict__ xs) {
  int b = blockIdx.y, y = blockIdx.x, t = threadIdx.x;  // 256 threads
  __shared__ unsigned short tile[64][72];               // [ci][x], padded
  for (int cb = 0; cb < 8; cb++) {
    int x = t & 63;
    int r0 = t >> 6;                                    // 0..3
#pragma unroll
    for (int it = 0; it < 16; it++) {
      int cr = r0 + it * 4;
      int ci = cb * 64 + cr;
      tile[cr][x] = f2bf(input[((b * CINC + ci) * HWDIM + y) * HWDIM + x] * style[b * CINC + ci]);
    }
    __syncthreads();
    int c2  = (t & 31) * 2;
    int xw0 = t >> 5;                                   // 0..7
#pragma unroll
    for (int it = 0; it < 8; it++) {
      int xw = xw0 + it * 8;
      unsigned int pack = (unsigned int)tile[c2][xw] | ((unsigned int)tile[c2 + 1][xw] << 16);
      *(unsigned int*)&xs[((b * XPAD + y + 1) * XPAD + xw + 1) * CINC + cb * 64 + c2] = pack;
    }
    __syncthreads();
  }
}

// ---------------- WMMA implicit-GEMM conv ----------------
// Tile: 128 cout x 64 pixels (one image row) per WG, 8 wave32s, each wave 16x64 (4 accs).
// K staged in chunks of 64 channels, double-buffered LDS, async global->LDS loads.
#define AROWS 128
#define BROWS 64
#define KSTG  64
#define LPITCH 72   // 64 elems + 8 pad -> 144B rows (16B-aligned, conflict-free)

__global__ __launch_bounds__(256) void k_conv(const unsigned short* __restrict__ xs,
                                              const unsigned short* __restrict__ wq,
                                              float* __restrict__ out) {
  int cb = blockIdx.x;          // cout block 0..3 (128 each)
  int y  = blockIdx.y;          // image row 0..63
  int b  = blockIdx.z;          // batch 0..7
  int t  = threadIdx.x;
  int lane = t & 31, w = t >> 5;       // w = M strip 0..7
  int ml   = lane & 15;
  int half = lane >> 4;

  __shared__ __attribute__((aligned(16))) unsigned short Ash[2][AROWS * LPITCH]; // [cout][k]
  __shared__ __attribute__((aligned(16))) unsigned short Bsh[2][BROWS * LPITCH]; // [pix][k]

  v8f acc[4];
#pragma unroll
  for (int i = 0; i < 4; i++) acc[i] = (v8f){};

  const unsigned short* wgbase = wq + (cb * AROWS) * CINC;
  const unsigned short* xgbase = xs + (b * XPAD + y) * XPAD * CINC;

  // ---- stage issue: 6 x 16B per thread (4 A segs + 2 B segs) ----
  auto issue = [&](int s) {
    int kk = s >> 3;                 // tap 0..8
    int st = s & 7;                  // 64-channel chunk 0..7
    int kh = (kk * 11) >> 5;         // kk/3 for kk in 0..8
    int kw = kk - kh * 3;
    int pb = s & 1;
    const unsigned short* Ag = wgbase + (kk * COUTC) * CINC + st * KSTG;
    const unsigned short* Bg = xgbase + (kh * XPAD + kw) * CINC + st * KSTG;
    unsigned short* Al = &Ash[pb][0];
    unsigned short* Bl = &Bsh[pb][0];
#pragma unroll
    for (int i = 0; i < 4; i++) {
      int sgi = t + 256 * i;         // 0..1023
      int row = sgi >> 3;            // 0..127
      int ko  = (sgi & 7) * 8;       // 0..56
      copy16(Al + row * LPITCH + ko, Ag + row * CINC + ko);
    }
#pragma unroll
    for (int i = 0; i < 2; i++) {
      int sgi = t + 256 * i;         // 0..511
      int p   = sgi >> 3;            // 0..63 (pixel x)
      int ko  = (sgi & 7) * 8;
      copy16(Bl + p * LPITCH + ko, Bg + p * CINC + ko);
    }
  };

  issue(0);

  for (int s = 0; s < 72; s++) {     // 9 taps * 8 chunks
    if (s < 71) {
      issue(s + 1);
#if HAVE_ASYNC
      __builtin_amdgcn_s_wait_asynccnt(6);   // stage-s data landed; 6 newest in flight
#endif
    } else {
#if HAVE_ASYNC
      __builtin_amdgcn_s_wait_asynccnt(0);
#endif
    }
    __syncthreads();                 // all waves' stage-s data visible

    const unsigned short* Ac = &Ash[s & 1][0];
    const unsigned short* Bc = &Bsh[s & 1][0];
    int afb = (w * 16 + ml) * LPITCH;
#pragma unroll
    for (int c2 = 0; c2 < 2; c2++) { // two 32-deep WMMA chunks per stage
      union { v16bf v; v8us h[2]; } au;
      au.h[0] = *(const v8us*)&Ac[afb + c2 * 32 + half * 8];
      au.h[1] = *(const v8us*)&Ac[afb + c2 * 32 + 16 + half * 8];
#pragma unroll
      for (int nt = 0; nt < 4; nt++) {
        union { v16bf v; v8us h[2]; } bu;
        int pbse = (nt * 16 + ml) * LPITCH + c2 * 32 + half * 8;
        bu.h[0] = *(const v8us*)&Bc[pbse];
        bu.h[1] = *(const v8us*)&Bc[pbse + 16];
        acc[nt] = __builtin_amdgcn_wmma_f32_16x16x32_bf16(
            false, au.v, false, bu.v, (short)0, acc[nt], false, false);
      }
    }
    __syncthreads();                 // stage-s buffer free for reuse at s+2
  }

  // C/D layout: VGPR r, lane l -> M = r + 8*(l>=16), N = l&15
  int mbase = cb * AROWS + w * 16 + half * 8;
  float* ob = out + ((b * COUTC + mbase) * HWDIM + y) * HWDIM + ml;
#pragma unroll
  for (int nt = 0; nt < 4; nt++) {
#pragma unroll
    for (int r = 0; r < 8; r++) {
      ob[r * (HWDIM * HWDIM) + nt * 16] = acc[nt][r];
    }
  }
}

// ---------------- host launcher ----------------
extern "C" void kernel_launch(void* const* d_in, const int* in_sizes, int n_in,
                              void* d_out, int out_size, void* d_ws, size_t ws_size,
                              hipStream_t stream) {
  const float* h1    = (const float*)d_in[0];
  const float* input = (const float*)d_in[1];
  const float* wsty  = (const float*)d_in[2];
  const float* cw    = (const float*)d_in[3];
  const float* w_ih  = (const float*)d_in[4];
  const float* w_hh  = (const float*)d_in[5];
  const float* b_ih  = (const float*)d_in[6];
  const float* b_hh  = (const float*)d_in[7];
  const float* aff_w = (const float*)d_in[8];
  const float* aff_b = (const float*)d_in[9];

  float* out_h2    = (float*)d_out;                            // 8*512
  float* out_conv  = out_h2 + BATCH * HIDC;                    // 8*512*64*64
  float* out_style = out_conv + BATCH * COUTC * HWDIM * HWDIM; // 8*512

  char* ws = (char*)d_ws;
  float* gx = (float*)ws;                                  //  49152 B
  float* gh = (float*)(ws + 49152);                        //  49152 B
  float* sv = (float*)(ws + 98304);                        //  16384 B
  unsigned short* wq = (unsigned short*)(ws + 114688);     // 9*512*512*2 = 4718592 B
  unsigned short* xs = (unsigned short*)(ws + 114688 + 4718592); // 8*66*66*512*2 B

  const int xs_n32 = (BATCH * XPAD * XPAD * CINC) / 2;     // ushorts -> uints

  k_gates<<<48, 256, 0, stream>>>(wsty, h1, w_ih, w_hh, b_ih, b_hh, gx, gh);
  k_h2   <<<16, 256, 0, stream>>>(gx, gh, h1, out_h2);
  k_s    <<<16, 256, 0, stream>>>(out_h2, aff_w, aff_b, sv);
  k_style<<<8, 256, 0, stream>>>(sv, out_style);
  k_wprep<<<1024, 256, 0, stream>>>(cw, wq);
  k_xzero<<<(xs_n32 + 255) / 256, 256, 0, stream>>>((unsigned int*)xs, xs_n32);
  k_xprep<<<dim3(64, 8), 256, 0, stream>>>(input, out_style, xs);
  k_conv <<<dim3(4, 64, 8), 256, 0, stream>>>(xs, wq, out_conv);
}
